// DeepseekV3DecoderLayer_52785148067901
// MI455X (gfx1250) — compile-verified, matrix operand order checked by reading, not run on hardware
//
#include <hip/hip_runtime.h>
#include <hip/hip_bf16.h>
#include <stdint.h>

// ---------------- problem constants ----------------
#define S_      1024
#define D_      4096
#define H_      32
#define QLORA_  1536
#define KVLORA_ 512
#define KV_A_   576          // KVLORA + DR
#define DN_     128
#define DR_     64
#define DQ_     192          // DN + DR
#define DV_     128
#define HDQ_    (H_ * DQ_)   // 6144
#define HDKV_   (H_ * 256)   // 8192
#define HV_     (H_ * DV_)   // 4096
#define INTER_  11008
#define EPS_    1e-6f
#define SCALE_  0.07216878364870322f   // (DN+DR)^-0.5 = 192^-0.5

typedef __bf16 bf16;
typedef __attribute__((ext_vector_type(16))) __bf16 v16bf;
typedef __attribute__((ext_vector_type(8)))  float  v8f;

// CDNA5 async global->LDS path (ASYNCcnt); signature probed in round 3.
#if defined(__AMDGCN__) && __has_builtin(__builtin_amdgcn_global_load_async_to_lds_b128)
#define ASYNC_LDS 1
typedef int v4i_ __attribute__((vector_size(16)));
typedef __attribute__((address_space(1))) v4i_ as1_v4i;
typedef __attribute__((address_space(3))) v4i_ as3_v4i;
#else
#define ASYNC_LDS 0
#endif

__device__ __forceinline__ void wait_async_lds() {
#if ASYNC_LDS
#if __has_builtin(__builtin_amdgcn_s_wait_asynccnt)
    __builtin_amdgcn_s_wait_asynccnt(0);
#else
    asm volatile("s_wait_asynccnt 0x0" ::: "memory");
#endif
#endif
}

// ---------------- small helpers ----------------
__device__ __forceinline__ bf16 f2bf(float f) {
    union { float f; unsigned u; } x; x.f = f;
    unsigned r = x.u + 0x7FFFu + ((x.u >> 16) & 1u);   // round-to-nearest-even
    unsigned short h = (unsigned short)(r >> 16);
    bf16 out; __builtin_memcpy(&out, &h, 2); return out;
}
// two floats -> packed bf16x2 dword (lo = a, hi = b)
__device__ __forceinline__ unsigned pack_bf16x2(float a, float b) {
#if defined(__AMDGCN__) && __has_builtin(__builtin_amdgcn_cvt_pk_bf16_f32)
    typedef __attribute__((ext_vector_type(2))) __bf16 v2bf;
    v2bf v = __builtin_amdgcn_cvt_pk_bf16_f32(a, b);
    unsigned u; __builtin_memcpy(&u, &v, 4); return u;
#else
    union { float f; unsigned u; } xa, xb; xa.f = a; xb.f = b;
    unsigned ra = xa.u + 0x7FFFu + ((xa.u >> 16) & 1u);
    unsigned rb = xb.u + 0x7FFFu + ((xb.u >> 16) & 1u);
    return (ra >> 16) | (rb & 0xFFFF0000u);
#endif
}
// K index held by element j of an A/B fragment for this half-wave (ISA 16-bit layout)
__device__ __forceinline__ int frag_k(int laneHi, int j) {
    return (j < 8 ? j : j + 8) + laneHi * 8;
}
__device__ __forceinline__ v8f wmma_bf16(v16bf a, v16bf b, v8f c) {
    return __builtin_amdgcn_wmma_f32_16x16x32_bf16(false, a, false, b, (short)0, c, false, false);
}
__device__ __forceinline__ float half_max(float x) {   // reduce over 16-lane half
    x = fmaxf(x, __shfl_xor(x, 1, 16));
    x = fmaxf(x, __shfl_xor(x, 2, 16));
    x = fmaxf(x, __shfl_xor(x, 4, 16));
    x = fmaxf(x, __shfl_xor(x, 8, 16));
    return x;
}
__device__ __forceinline__ float half_sum(float x) {
    x += __shfl_xor(x, 1, 16);
    x += __shfl_xor(x, 2, 16);
    x += __shfl_xor(x, 4, 16);
    x += __shfl_xor(x, 8, 16);
    return x;
}

// ---------------- RMSNorm (fp32 in -> bf16 out) ----------------
__global__ void rmsnorm_bf16_k(const float* __restrict__ X, const float* __restrict__ W,
                               bf16* __restrict__ Y, int cols, int istride, int ostride) {
    const int row = blockIdx.x;
    const float* x = X + (size_t)row * istride;
    bf16* y = Y + (size_t)row * ostride;
    float ss = 0.f;
    for (int c = threadIdx.x; c < cols; c += blockDim.x) { float v = x[c]; ss += v * v; }
    __shared__ float red[256];
    red[threadIdx.x] = ss; __syncthreads();
    for (int s = 128; s > 0; s >>= 1) {
        if ((int)threadIdx.x < s) red[threadIdx.x] += red[threadIdx.x + s];
        __syncthreads();
    }
    float inv = rsqrtf(red[0] / (float)cols + EPS_);
    for (int c = threadIdx.x; c < cols; c += blockDim.x) y[c] = f2bf(x[c] * inv * W[c]);
}

// ---------------- LDS-staged WMMA GEMM (ping-pong double buffer) ----------------
// C(f32, MxN) = A(bf16, MxK) @ B(f32 weights, KxN)
// Block tile 128x128, 8 waves of 64x32 (8 WMMAs / wave / 32-k step).
// A tile: async global->LDS b128 (ASYNCcnt). B tile: incremental-pointer
// coalesced b32 loads, bf16x2 pack, ds_store_b32, stored TRANSPOSED [col][k].
// Stage buffer `nxt` while computing from `cur`; one wait+barrier per step.
__global__ void __launch_bounds__(256)
gemm_bf16w_k(const bf16* __restrict__ A, const float* __restrict__ B,
             float* __restrict__ C, int M, int N, int K) {
    const int t    = threadIdx.x;
    const int lane = t & 31;
    const int wave = t >> 5;
    const int hi   = lane >> 4;
    const int n    = lane & 15;
    const int mw   = wave >> 2;          // 0..1 : 64-row slice
    const int nw   = wave & 3;           // 0..3 : 32-col slice
    const int tileM = blockIdx.x * 128;
    const int tileN = blockIdx.y * 128;

    __shared__ bf16 As[2][128 * 32];     // [row][k]   2 x 8 KB
    __shared__ bf16 Bs[2][128 * 32];     // [col][k]   2 x 8 KB (transposed)

    v8f acc[4][2];
#pragma unroll
    for (int mt = 0; mt < 4; ++mt)
#pragma unroll
        for (int nt = 0; nt < 2; ++nt)
#pragma unroll
            for (int r = 0; r < 8; ++r) acc[mt][nt][r] = 0.f;

    // ---- loop-invariant staging coordinates ----
    // A: two 16B chunks per thread
    int arow[2], akc[2];
    const bf16* abase[2];
#pragma unroll
    for (int i = 0; i < 2; ++i) {
        const int c = i * 256 + t;                   // 0..511
        arow[i] = c >> 2;                            // 0..127
        akc[i]  = (c & 3) * 8;                       // 0,8,16,24
        abase[i] = A + (size_t)(tileM + arow[i]) * K + akc[i];
    }
    // B: one column per thread, k-pairs packed to dwords; 16 pointers bumped by
    // 32*N per step so there are NO multiplies in the loop.
    const int colB  = t & 127;                       // 0..127
    const int kp0   = t >> 7;                        // 0..1
    const int bcol  = tileN + colB;
    const int bcolc = bcol < N ? bcol : N - 1;       // clamp loads, guard stores
    const size_t bstep = (size_t)N * 32;
    const float* bp0[8];
    const float* bp1[8];
#pragma unroll
    for (int i = 0; i < 8; ++i) {
        bp0[i] = B + (size_t)((kp0 + i * 2) * 2) * N + bcolc;
        bp1[i] = bp0[i] + N;
    }

    auto stageA = [&](int buf, int k0) {
#pragma unroll
        for (int i = 0; i < 2; ++i) {
            const bf16* gsrc = abase[i] + k0;
            bf16* ldst = &As[buf][arow[i] * 32 + akc[i]];
#if ASYNC_LDS
            __builtin_amdgcn_global_load_async_to_lds_b128(
                (as1_v4i*)(v4i_*)gsrc, (as3_v4i*)(v4i_*)ldst, 0, 0);
#else
            *(uint4*)ldst = *(const uint4*)gsrc;
#endif
        }
    };
    auto stageB = [&](int buf) {
        float e0[8], e1[8];
#pragma unroll
        for (int i = 0; i < 8; ++i) {
            e0[i] = *bp0[i]; bp0[i] += bstep;
            e1[i] = *bp1[i]; bp1[i] += bstep;
        }
#pragma unroll
        for (int i = 0; i < 8; ++i)
            *(unsigned*)&Bs[buf][colB * 32 + (kp0 + i * 2) * 2] = pack_bf16x2(e0[i], e1[i]);
    };

    // ---- prologue: fill buffer 0 ----
    stageA(0, 0);
    stageB(0);
    wait_async_lds();
    __syncthreads();

    int cur = 0;
    for (int k0 = 0; k0 < K; k0 += 32) {
        const int nxt = cur ^ 1;
        if (k0 + 32 < K) {               // stage next tile while computing this one
            stageA(nxt, k0 + 32);
            stageB(nxt);
        }
        // ---- fragments from LDS and 8 WMMAs ----
        v16bf af[4];
#pragma unroll
        for (int mt = 0; mt < 4; ++mt) {
            const int row = mw * 64 + mt * 16 + n;
#pragma unroll
            for (int j = 0; j < 16; ++j) af[mt][j] = As[cur][row * 32 + frag_k(hi, j)];
        }
        v16bf bfr[2];
#pragma unroll
        for (int nt = 0; nt < 2; ++nt) {
            const int col = nw * 32 + nt * 16 + n;
#pragma unroll
            for (int j = 0; j < 16; ++j) bfr[nt][j] = Bs[cur][col * 32 + frag_k(hi, j)];
        }
#pragma unroll
        for (int mt = 0; mt < 4; ++mt)
#pragma unroll
            for (int nt = 0; nt < 2; ++nt)
                acc[mt][nt] = wmma_bf16(af[mt], bfr[nt], acc[mt][nt]);

        wait_async_lds();                // next tile landed
        __syncthreads();                 // everyone done reading `cur`
        cur = nxt;
    }

    // ---- epilogue: C-layout store (lane half hi -> rows +8), column-guarded
#pragma unroll
    for (int mt = 0; mt < 4; ++mt) {
        const int mbase = tileM + mw * 64 + mt * 16 + hi * 8;
#pragma unroll
        for (int nt = 0; nt < 2; ++nt) {
            const int col = tileN + nw * 32 + nt * 16 + n;
            if (col < N) {
#pragma unroll
                for (int r = 0; r < 8; ++r)
                    C[(size_t)(mbase + r) * N + col] = acc[mt][nt][r];
            }
        }
    }
}

// ---------------- RoPE (in-place on q f32 and k_pe part of lkv) ----------------
__global__ void rope_k(float* __restrict__ q, float* __restrict__ lkv,
                       const float* __restrict__ cosb, const float* __restrict__ sinb) {
    const int s = blockIdx.x;
    const float* cr = cosb + (size_t)s * DR_;
    const float* sr = sinb + (size_t)s * DR_;
    for (int idx = threadIdx.x; idx < H_ * 32 + 32; idx += blockDim.x) {
        float* base; int i;
        if (idx < H_ * 32) { int h = idx >> 5; i = idx & 31; base = q + (size_t)s * HDQ_ + h * DQ_ + DN_; }
        else               { i = idx - H_ * 32;              base = lkv + (size_t)s * KV_A_ + KVLORA_; }
        const float c = cr[i], sn = sr[i];       // cos[i]==cos[i+32], sin[i]==sin[i+32]
        const float a = base[i], b = base[i + 32];
        base[i]      = a * c - b * sn;
        base[i + 32] = b * c + a * sn;
    }
}

// ---------------- assembly / elementwise ----------------
__global__ void conv_bf16_k(const float* __restrict__ X, unsigned* __restrict__ Y, size_t npair) {
    for (size_t i = blockIdx.x * (size_t)blockDim.x + threadIdx.x; i < npair;
         i += (size_t)gridDim.x * blockDim.x)
        Y[i] = pack_bf16x2(X[2 * i], X[2 * i + 1]);
}
__global__ void assemble_kv_k(const float* __restrict__ kv, const float* __restrict__ lkv,
                              bf16* __restrict__ Kb, bf16* __restrict__ Vb) {
    const int s = blockIdx.x;
    const float* kvr = kv  + (size_t)s * HDKV_;
    const float* pe  = lkv + (size_t)s * KV_A_ + KVLORA_;    // roped k_pe (64)
    bf16* kr = Kb + (size_t)s * (H_ * DQ_);
    bf16* vr = Vb + (size_t)s * (H_ * DV_);
    for (int idx = threadIdx.x; idx < H_ * DQ_; idx += blockDim.x) {
        const int h = idx / DQ_, d = idx - h * DQ_;
        kr[idx] = f2bf(d < DN_ ? kvr[h * 256 + d] : pe[d - DN_]);
    }
    for (int idx = threadIdx.x; idx < H_ * DV_; idx += blockDim.x) {
        const int h = idx / DV_, d = idx - h * DV_;
        vr[idx] = f2bf(kvr[h * 256 + DN_ + d]);
    }
}
__global__ void add_k(const float* __restrict__ A, const float* __restrict__ B,
                      float* __restrict__ C, size_t n) {
    for (size_t i = blockIdx.x * (size_t)blockDim.x + threadIdx.x; i < n;
         i += (size_t)gridDim.x * blockDim.x) C[i] = A[i] + B[i];
}
__global__ void silu_mul_k(const float* __restrict__ G, const float* __restrict__ U,
                           unsigned* __restrict__ Y, size_t npair) {
    for (size_t i = blockIdx.x * (size_t)blockDim.x + threadIdx.x; i < npair;
         i += (size_t)gridDim.x * blockDim.x) {
        const float g0 = G[2 * i], g1 = G[2 * i + 1];
        const float a = (g0 / (1.f + __expf(-g0))) * U[2 * i];
        const float b = (g1 / (1.f + __expf(-g1))) * U[2 * i + 1];
        Y[i] = pack_bf16x2(a, b);
    }
}

// ---------------- flash attention (MLA), one wave per (head, 16-query tile) --------
__global__ void __launch_bounds__(32)
mla_flash_k(const bf16* __restrict__ Q, const bf16* __restrict__ K,
            const bf16* __restrict__ V, bf16* __restrict__ O) {
    const int h     = blockIdx.y;
    const int qBase = blockIdx.x * 16;
    const int lane  = threadIdx.x & 31;
    const int hi    = lane >> 4;
    const int n     = lane & 15;
    const int ldq   = H_ * DQ_;
    const int ldv   = H_ * DV_;

    const bf16* qrow = Q + (size_t)(qBase + n) * ldq + h * DQ_;
    v16bf qa[6];
#pragma unroll
    for (int tt = 0; tt < 6; ++tt)
#pragma unroll
        for (int j = 0; j < 16; ++j) qa[tt][j] = qrow[tt * 32 + frag_k(hi, j)];

    v8f acc[8];
    float mrun[8], lrun[8];
#pragma unroll
    for (int g = 0; g < 8; ++g)
#pragma unroll
        for (int r = 0; r < 8; ++r) acc[g][r] = 0.f;
#pragma unroll
    for (int r = 0; r < 8; ++r) { mrun[r] = -__builtin_inff(); lrun[r] = 0.f; }

    __shared__ bf16 P[16 * 32];                  // one wave per block
    const int jbMax = (qBase + 15) >> 5;
    for (int jb = 0; jb <= jbMax; ++jb) {
        const int kb = jb * 32;
        v8f s0 = {0.f,0.f,0.f,0.f,0.f,0.f,0.f,0.f};
        v8f s1 = {0.f,0.f,0.f,0.f,0.f,0.f,0.f,0.f};
        const bf16* k0row = K + (size_t)(kb + n) * ldq + h * DQ_;
        const bf16* k1row = k0row + (size_t)16 * ldq;
#pragma unroll
        for (int tt = 0; tt < 6; ++tt) {
            v16bf b0, b1;
#pragma unroll
            for (int j = 0; j < 16; ++j) {
                const int kk = tt * 32 + frag_k(hi, j);
                b0[j] = k0row[kk];
                b1[j] = k1row[kk];
            }
            s0 = wmma_bf16(qa[tt], b0, s0);
            s1 = wmma_bf16(qa[tt], b1, s1);
        }
#pragma unroll
        for (int r = 0; r < 8; ++r) {
            const int qrg = qBase + hi * 8 + r;
            float v0 = (kb + n      <= qrg) ? s0[r] * SCALE_ : -__builtin_inff();
            float v1 = (kb + 16 + n <= qrg) ? s1[r] * SCALE_ : -__builtin_inff();
            const float mx   = half_max(fmaxf(v0, v1));
            const float mnew = fmaxf(mrun[r], mx);
            const float corr = __expf(mrun[r] - mnew);
            const float p0 = __expf(v0 - mnew);
            const float p1 = __expf(v1 - mnew);
            lrun[r] = lrun[r] * corr + half_sum(p0 + p1);
            mrun[r] = mnew;
#pragma unroll
            for (int g = 0; g < 8; ++g) acc[g][r] *= corr;
            const int mr = hi * 8 + r;
            P[mr * 32 + n]      = f2bf(p0);
            P[mr * 32 + 16 + n] = f2bf(p1);
        }
        v16bf pa;
#pragma unroll
        for (int j = 0; j < 16; ++j) pa[j] = P[n * 32 + frag_k(hi, j)];
#pragma unroll
        for (int g = 0; g < 8; ++g) {
            v16bf vb;
#pragma unroll
            for (int j = 0; j < 16; ++j)
                vb[j] = V[(size_t)(kb + frag_k(hi, j)) * ldv + h * DV_ + g * 16 + n];
            acc[g] = wmma_bf16(pa, vb, acc[g]);
        }
    }
#pragma unroll
    for (int r = 0; r < 8; ++r) {
        const int row = qBase + hi * 8 + r;
        const float invl = 1.f / lrun[r];
#pragma unroll
        for (int g = 0; g < 8; ++g)
            O[(size_t)row * (H_ * DV_) + h * DV_ + g * 16 + n] = f2bf(acc[g][r] * invl);
    }
}

// ---------------- host-side orchestration ----------------
extern "C" void kernel_launch(void* const* d_in, const int* in_sizes, int n_in,
                              void* d_out, int out_size, void* d_ws, size_t ws_size,
                              hipStream_t stream) {
    const float* hidden_states = (const float*)d_in[0];
    const float* input_ln_w    = (const float*)d_in[1];
    const float* q_a_w         = (const float*)d_in[2];
    const float* q_a_ln_w      = (const float*)d_in[3];
    const float* q_b_w         = (const float*)d_in[4];
    const float* kv_a_w        = (const float*)d_in[5];
    const float* kv_a_ln_w     = (const float*)d_in[6];
    const float* kv_b_w        = (const float*)d_in[7];
    const float* o_w           = (const float*)d_in[8];
    const float* post_ln_w     = (const float*)d_in[9];
    const float* gate_w        = (const float*)d_in[10];
    const float* up_w          = (const float*)d_in[11];
    const float* down_w        = (const float*)d_in[12];
    const float* cosb          = (const float*)d_in[13];
    const float* sinb          = (const float*)d_in[14];
    float* out = (float*)d_out;

    char* p = (char*)d_ws;
    auto alloc = [&](size_t bytes) -> void* {
        void* r = (void*)p; p += (bytes + 255) & ~(size_t)255; return r;
    };
    bf16*  x_bf  = (bf16*)alloc((size_t)S_ * D_ * 2);
    float* lq    = (float*)alloc((size_t)S_ * QLORA_ * 4);
    bf16*  lqn   = (bf16*)alloc((size_t)S_ * QLORA_ * 2);
    float* q     = (float*)alloc((size_t)S_ * HDQ_ * 4);
    float* lkv   = (float*)alloc((size_t)S_ * KV_A_ * 4);
    bf16*  kvn   = (bf16*)alloc((size_t)S_ * KVLORA_ * 2);
    float* kv    = (float*)alloc((size_t)S_ * HDKV_ * 4);
    bf16*  Qb    = (bf16*)alloc((size_t)S_ * HDQ_ * 2);
    bf16*  Kb    = (bf16*)alloc((size_t)S_ * H_ * DQ_ * 2);
    bf16*  Vb    = (bf16*)alloc((size_t)S_ * H_ * DV_ * 2);
    bf16*  attnb = (bf16*)alloc((size_t)S_ * HV_ * 2);
    float* oproj = (float*)alloc((size_t)S_ * D_ * 4);
    float* hid   = (float*)alloc((size_t)S_ * D_ * 4);
    bf16*  x2b   = (bf16*)alloc((size_t)S_ * D_ * 2);
    float* gbuf  = (float*)alloc((size_t)S_ * INTER_ * 4);
    float* ubuf  = (float*)alloc((size_t)S_ * INTER_ * 4);
    bf16*  hm    = (bf16*)alloc((size_t)S_ * INTER_ * 2);
    float* mlp   = (float*)alloc((size_t)S_ * D_ * 4);
    (void)ws_size; (void)in_sizes; (void)n_in; (void)out_size;

    const dim3 blk256(256), blk32(32);
    auto gemm_grid = [](int N) { return dim3(S_ / 128, (N + 127) / 128); };

    hipLaunchKernelGGL(rmsnorm_bf16_k, dim3(S_), blk256, 0, stream,
                       hidden_states, input_ln_w, x_bf, D_, D_, D_);
    hipLaunchKernelGGL(gemm_bf16w_k, gemm_grid(QLORA_), blk256, 0, stream,
                       x_bf, q_a_w, lq, S_, QLORA_, D_);
    hipLaunchKernelGGL(rmsnorm_bf16_k, dim3(S_), blk256, 0, stream,
                       lq, q_a_ln_w, lqn, QLORA_, QLORA_, QLORA_);
    hipLaunchKernelGGL(gemm_bf16w_k, gemm_grid(HDQ_), blk256, 0, stream,
                       lqn, q_b_w, q, S_, HDQ_, QLORA_);
    hipLaunchKernelGGL(gemm_bf16w_k, gemm_grid(KV_A_), blk256, 0, stream,
                       x_bf, kv_a_w, lkv, S_, KV_A_, D_);
    hipLaunchKernelGGL(rmsnorm_bf16_k, dim3(S_), blk256, 0, stream,
                       lkv, kv_a_ln_w, kvn, KVLORA_, KV_A_, KVLORA_);
    hipLaunchKernelGGL(rope_k, dim3(S_), blk256, 0, stream, q, lkv, cosb, sinb);
    hipLaunchKernelGGL(gemm_bf16w_k, gemm_grid(HDKV_), blk256, 0, stream,
                       kvn, kv_b_w, kv, S_, HDKV_, KVLORA_);
    hipLaunchKernelGGL(conv_bf16_k, dim3(2048), blk256, 0, stream,
                       q, (unsigned*)Qb, (size_t)S_ * HDQ_ / 2);
    hipLaunchKernelGGL(assemble_kv_k, dim3(S_), blk256, 0, stream, kv, lkv, Kb, Vb);
    hipLaunchKernelGGL(mla_flash_k, dim3(S_ / 16, H_), blk32, 0, stream, Qb, Kb, Vb, attnb);
    hipLaunchKernelGGL(gemm_bf16w_k, gemm_grid(D_), blk256, 0, stream,
                       attnb, o_w, oproj, S_, D_, HV_);
    hipLaunchKernelGGL(add_k, dim3(2048), blk256, 0, stream,
                       hidden_states, oproj, hid, (size_t)S_ * D_);
    hipLaunchKernelGGL(rmsnorm_bf16_k, dim3(S_), blk256, 0, stream,
                       hid, post_ln_w, x2b, D_, D_, D_);
    hipLaunchKernelGGL(gemm_bf16w_k, gemm_grid(INTER_), blk256, 0, stream,
                       x2b, gate_w, gbuf, S_, INTER_, D_);
    hipLaunchKernelGGL(gemm_bf16w_k, gemm_grid(INTER_), blk256, 0, stream,
                       x2b, up_w, ubuf, S_, INTER_, D_);
    hipLaunchKernelGGL(silu_mul_k, dim3(2048), blk256, 0, stream,
                       gbuf, ubuf, (unsigned*)hm, (size_t)S_ * INTER_ / 2);
    hipLaunchKernelGGL(gemm_bf16w_k, gemm_grid(D_), blk256, 0, stream,
                       hm, down_w, mlp, S_, D_, INTER_);
    hipLaunchKernelGGL(add_k, dim3(2048), blk256, 0, stream,
                       hid, mlp, out, (size_t)S_ * D_);
}
